// FlaxFalconBlock_11450382811921
// MI455X (gfx1250) — compile-verified, matrix operand order checked by reading, not run on hardware
//
#include <hip/hip_runtime.h>
#include <hip/hip_bf16.h>
#include <math.h>

// ---------------- problem constants ----------------
#define B_    2
#define S_    1024
#define H_    4096
#define NH_   32
#define NKV_  8
#define HD_   128
#define G_    4
#define QKV_  6144      // (NKV*2 + NH) * HD
#define FF_   16384
#define M2_   2048      // B*S
#define EPS_  1e-5f
#define NEGBIG (-1.0e30f)

typedef __bf16 bf16_t;
typedef bf16_t v16bf __attribute__((ext_vector_type(16)));
typedef float  v8f   __attribute__((ext_vector_type(8)));
typedef unsigned int v4u __attribute__((ext_vector_type(4)));
typedef int v8i __attribute__((ext_vector_type(8)));
typedef int v4i __attribute__((ext_vector_type(4)));

union Frag16 { v16bf v; unsigned u[8]; unsigned short s[16]; };

__device__ __forceinline__ unsigned short f2bf(float f) {
  unsigned u = __float_as_uint(f);
  u += 0x7fffu + ((u >> 16) & 1u);          // round-to-nearest-even
  return (unsigned short)(u >> 16);
}

// K index pattern for 16-bit A/B fragments (CDNA5 ISA 7.12.2):
// VGPR j of lane (row, half): j<4 -> K = 2j + 8*half ; j>=4 -> K = 16 + 2(j-4) + 8*half
__device__ __forceinline__ int kpat(int j, int half) {
  return ((j < 4) ? (2 * j) : (16 + 2 * (j - 4))) + 8 * half;
}

__device__ __forceinline__ v8f vzero8() {
  v8f z = {0.f, 0.f, 0.f, 0.f, 0.f, 0.f, 0.f, 0.f};
  return z;
}

// ---------------- Tensor Data Mover: 2-D bf16 tile -> LDS ----------------
// tile_w elems/row, tile_h rows, global row stride stride0 (elems).
// LDS padding: after every (pad_interval) dwords insert (pad_amount) dwords.
//   pad_int_code: 3 -> 16 DW (=32 bf16), 5 -> 64 DW (=128 bf16)
//   pad_amt_code: 0 -> 1 DW (=2 bf16),   3 -> 4 DW (=8 bf16)
__device__ __forceinline__ void tdm_load_2d_bf16(const unsigned short* g,
                                                 unsigned lds_off,
                                                 unsigned tile_w, unsigned tile_h,
                                                 unsigned long tensor_w,
                                                 unsigned long stride0,
                                                 int pad_int_code, int pad_amt_code) {
  const unsigned long ga = (unsigned long)g;
  v4u g0;
  g0[0] = 1u;                                             // count=1, user mode
  g0[1] = lds_off;                                        // LDS byte address
  g0[2] = (unsigned)ga;                                   // global addr [31:0]
  g0[3] = (unsigned)((ga >> 32) & 0x01ffffffu) | (2u << 30); // addr[56:32] | type=2
  v8i g1;
  g1[0] = (1 << 16)                                       // data_size = 2 bytes
        | (1 << 20)                                       // pad_enable
        | (pad_int_code << 22) | (pad_amt_code << 25);
  g1[1] = (int)((tensor_w & 0xffffu) << 16);              // tensor_dim0[15:0]
  g1[2] = (int)(((tensor_w >> 16) & 0xffffu)              // tensor_dim0[31:16]
        | ((unsigned)tile_h << 16));                      // tensor_dim1[15:0]
  g1[3] = (int)(tile_w << 16);                            // tile_dim0
  g1[4] = (int)tile_h;                                    // tile_dim1 (tile_dim2=0)
  g1[5] = (int)(stride0 & 0xffffffffu);                   // tensor_dim0_stride[31:0]
  g1[6] = (int)((stride0 >> 32) & 0xffffu);               // stride0[47:32], dim1_stride lo=0
  g1[7] = 0;
  v4i zz4 = {0, 0, 0, 0};
  v8i zz8 = {0, 0, 0, 0, 0, 0, 0, 0};
  __builtin_amdgcn_tensor_load_to_lds(g0, g1, zz4, zz4, zz8, 0);
}

// ---------------- f32 -> bf16 conversion ----------------
__global__ void convert_bf16_kernel(const float* __restrict__ in,
                                    unsigned short* __restrict__ out, long n4) {
  long i = (long)blockIdx.x * blockDim.x + threadIdx.x;
  long stride = (long)gridDim.x * blockDim.x;
  for (; i < n4; i += stride) {
    float4 f = ((const float4*)in)[i];
    ushort4 o;
    o.x = f2bf(f.x); o.y = f2bf(f.y); o.z = f2bf(f.z); o.w = f2bf(f.w);
    ((ushort4*)out)[i] = o;
  }
}

// ---------------- dual LayerNorm (attn + mlp), bf16 outputs ----------------
__global__ __launch_bounds__(256) void ln_dual_kernel(
    const float* __restrict__ x,
    const float* __restrict__ sA, const float* __restrict__ bA,
    const float* __restrict__ sM, const float* __restrict__ bM,
    unsigned short* __restrict__ outA, unsigned short* __restrict__ outM) {
  __shared__ float red[256];
  const int rowi = blockIdx.x;
  const float* xr = x + (size_t)rowi * H_;
  float v[16];
  float sum = 0.f;
#pragma unroll
  for (int i = 0; i < 16; ++i) { v[i] = xr[threadIdx.x + i * 256]; sum += v[i]; }
  red[threadIdx.x] = sum; __syncthreads();
  for (int off = 128; off > 0; off >>= 1) {
    if (threadIdx.x < off) red[threadIdx.x] += red[threadIdx.x + off];
    __syncthreads();
  }
  const float mean = red[0] * (1.0f / H_);
  __syncthreads();
  float s2 = 0.f;
#pragma unroll
  for (int i = 0; i < 16; ++i) { float d = v[i] - mean; s2 += d * d; }
  red[threadIdx.x] = s2; __syncthreads();
  for (int off = 128; off > 0; off >>= 1) {
    if (threadIdx.x < off) red[threadIdx.x] += red[threadIdx.x + off];
    __syncthreads();
  }
  const float rstd = rsqrtf(red[0] * (1.0f / H_) + EPS_);
#pragma unroll
  for (int i = 0; i < 16; ++i) {
    const int c = threadIdx.x + i * 256;
    const float xn = (v[i] - mean) * rstd;
    outA[(size_t)rowi * H_ + c] = f2bf(xn * sA[c] + bA[c]);
    outM[(size_t)rowi * H_ + c] = f2bf(xn * sM[c] + bM[c]);
  }
}

// ---------------- 128x128 WMMA bf16 GEMM, f32 accumulate ----------------
// A tile staged by TDM (tensor_load_to_lds, padded), B tile pair-pack transposed.
// EPI: 0 = store f32, 1 = gelu(x) -> bf16, 2 = store f32 of (c + r1 + r2)
template <int EPI>
__global__ __launch_bounds__(256) void gemm_bf16_kernel(
    const unsigned short* __restrict__ A,   // [M x K] bf16 row-major
    const unsigned short* __restrict__ Bw,  // [K x N] bf16 row-major
    int N, int K,
    float* __restrict__ Cf, unsigned short* __restrict__ Cb,
    const float* __restrict__ r1, const float* __restrict__ r2) {
  __shared__ unsigned short As[128 * 34];   // [m][k], stride 34 (TDM pad)
  __shared__ unsigned short Bs[128 * 34];   // transposed: [n][k], stride 34
  const int tid  = threadIdx.x;
  const int w    = tid >> 5, lane = tid & 31;
  const int wu   = __builtin_amdgcn_readfirstlane(w);   // wave-uniform wave id
  const int row  = lane & 15, half = lane >> 4;
  const int mo   = (w & 3) * 32;            // wave tile: 32 (M) x 64 (N)
  const int no   = (w >> 2) * 64;
  const int bm   = blockIdx.y, bn = blockIdx.x;
  const unsigned As_off = (unsigned)(unsigned long long)(void*)As;

  v8f acc[2][4];
#pragma unroll
  for (int i = 0; i < 2; ++i)
#pragma unroll
    for (int j = 0; j < 4; ++j) acc[i][j] = vzero8();

  // B staging: 16 k-pairs x 16 n-groups of 8; pack 2 k-rows into u32
  const int kp2 = (tid >> 4) * 2;           // even k row (0..30)
  const int n0  = (tid & 15) * 8;
  const unsigned short* gB = Bw + (size_t)kp2 * N + bn * 128 + n0;
  const unsigned short* gAt = A + (size_t)(bm * 128) * K;

  const int kiters = K >> 5;
  for (int it = 0; it < kiters; ++it) {
    __syncthreads();
    if (wu == 0) {    // one wave issues the TDM copy of the A tile
      tdm_load_2d_bf16(gAt + it * 32, As_off, /*tile_w=*/32, /*tile_h=*/128,
                       (unsigned long)K, (unsigned long)K,
                       /*pad 16DW*/3, /*+1DW*/0);
    }
    {   // B tile -> LDS transposed, pair-packed
      union { uint4 q; unsigned short s[8]; } lo, hi;
      const unsigned short* p = gB + (size_t)it * 32 * N;
      lo.q = *(const uint4*)p;
      hi.q = *(const uint4*)(p + N);
#pragma unroll
      for (int e = 0; e < 8; ++e)
        *(unsigned*)(Bs + (n0 + e) * 34 + kp2) =
            (unsigned)lo.s[e] | ((unsigned)hi.s[e] << 16);
    }
    if (wu == 0) __builtin_amdgcn_s_wait_tensorcnt(0);
    __syncthreads();

    Frag16 af[2], bfr[4];
#pragma unroll
    for (int mi = 0; mi < 2; ++mi)
#pragma unroll
      for (int j = 0; j < 8; ++j)
        af[mi].u[j] = *(const unsigned*)(As + (mo + mi * 16 + row) * 34 + kpat(j, half));
#pragma unroll
    for (int nj = 0; nj < 4; ++nj)
#pragma unroll
      for (int j = 0; j < 8; ++j)
        bfr[nj].u[j] = *(const unsigned*)(Bs + (no + nj * 16 + row) * 34 + kpat(j, half));
#pragma unroll
    for (int mi = 0; mi < 2; ++mi)
#pragma unroll
      for (int nj = 0; nj < 4; ++nj)
        acc[mi][nj] = __builtin_amdgcn_wmma_f32_16x16x32_bf16(
            false, af[mi].v, false, bfr[nj].v, (short)0, acc[mi][nj], false, false);
  }

#pragma unroll
  for (int mi = 0; mi < 2; ++mi)
#pragma unroll
    for (int nj = 0; nj < 4; ++nj)
#pragma unroll
      for (int e = 0; e < 8; ++e) {
        const int r = bm * 128 + mo + mi * 16 + e + 8 * half;
        const int c = bn * 128 + no + nj * 16 + row;
        const size_t idx = (size_t)r * N + c;
        const float cv = acc[mi][nj][e];
        if (EPI == 0) {
          Cf[idx] = cv;
        } else if (EPI == 1) {
          Cb[idx] = f2bf(0.5f * cv * (1.0f + erff(cv * 0.70710678118654752f)));
        } else {
          Cf[idx] = cv + r1[idx] + r2[idx];
        }
      }
}

// ---------------- RoPE + grouped head split, bf16 Q/K/V ----------------
__global__ __launch_bounds__(128) void rope_split_kernel(
    const float* __restrict__ qkv, const float* __restrict__ sin_tab,
    const float* __restrict__ cos_tab, const int* __restrict__ pos_ids,
    unsigned short* __restrict__ Qo, unsigned short* __restrict__ Ko,
    unsigned short* __restrict__ Vo) {
  const int bs = blockIdx.x;
  const int b = bs >> 10, s = bs & (S_ - 1);
  const int d = threadIdx.x;
  const int d2 = d ^ 64;
  const int pos = pos_ids[bs];
  const float cs = cos_tab[pos * HD_ + d], sn = sin_tab[pos * HD_ + d];
  const float sgn = (d < 64) ? -1.0f : 1.0f;   // rotate_half = [-x2, x1]
  const float* rowp = qkv + (size_t)bs * QKV_;
#pragma unroll
  for (int kv = 0; kv < NKV_; ++kv) {
    const float* base = rowp + kv * (G_ + 2) * HD_;
    const size_t kvrow = ((size_t)(b * NKV_ + kv) * S_ + s) * HD_;
    float xk = base[G_ * HD_ + d], xk2 = base[G_ * HD_ + d2];
    Ko[kvrow + d] = f2bf(xk * cs + sgn * xk2 * sn);
    Vo[kvrow + d] = f2bf(base[(G_ + 1) * HD_ + d]);
#pragma unroll
    for (int g = 0; g < G_; ++g) {
      const int h = kv * G_ + g;
      float xq = base[g * HD_ + d], xq2 = base[g * HD_ + d2];
      Qo[((size_t)(b * NH_ + h) * S_ + s) * HD_ + d] = f2bf(xq * cs + sgn * xq2 * sn);
    }
  }
}

// ---------------- flash attention: 64 queries/WG, 32-key blocks ----------------
// K tile staged by TDM (natural [key][hd], padded stride 136).
// V tile staged pair-pack transposed Vt[hd][key] so P*V B-frags are b128 reads.
__global__ __launch_bounds__(128) void flash_attn_kernel(
    const unsigned short* __restrict__ Q, const unsigned short* __restrict__ K,
    const unsigned short* __restrict__ V, const int* __restrict__ amask,
    unsigned short* __restrict__ ctx) {
  __shared__ unsigned short Ks[32 * 136];   // [key][hd], stride 136 (TDM pad)
  __shared__ unsigned short Vt[128 * 40];   // transposed: [hd][key], stride 40
  __shared__ unsigned short Ps[4 * 16 * 40];// per-wave P staging [m][k], stride 40
  const int bh = blockIdx.x;
  const int b = bh >> 5, h = bh & 31;
  const int kv = h >> 2;
  const int tid = threadIdx.x;
  const int w = tid >> 5, lane = tid & 31;
  const int wu = __builtin_amdgcn_readfirstlane(w);
  const int row = lane & 15, half = lane >> 4;
  const int q0wg = blockIdx.y * 64;
  const int qbase = q0wg + w * 16;
  const float scale = 0.08838834764831845f;   // 1/sqrt(128)
  const unsigned Ks_off = (unsigned)(unsigned long long)(void*)Ks;

  // Q fragments: 16 rows x 128 hd, 4 K-steps of 32
  Frag16 qf[4];
  const unsigned short* qrow = Q + ((size_t)(b * NH_ + h) * S_ + qbase + row) * HD_;
#pragma unroll
  for (int f = 0; f < 4; ++f)
#pragma unroll
    for (int j = 0; j < 8; ++j)
      qf[f].u[j] = *(const unsigned*)(qrow + f * 32 + kpat(j, half));

  float mrow[8], lrow[8];
  v8f oacc[8];
#pragma unroll
  for (int e = 0; e < 8; ++e) { mrow[e] = NEGBIG; lrow[e] = 0.f; }
#pragma unroll
  for (int t = 0; t < 8; ++t) oacc[t] = vzero8();

  const unsigned short* Kb = K + (size_t)(b * NKV_ + kv) * S_ * HD_;
  const unsigned short* Vb = V + (size_t)(b * NKV_ + kv) * S_ * HD_;
  const int nblk = (q0wg + 64) >> 5;          // causal extent for this WG
  // V staging: 16 key-pairs x 8 hd-groups of 16; pack 2 keys into u32
  const int vkey2 = (tid & 15) * 2;
  const int vc0 = (tid >> 4) * 16;

  for (int blk = 0; blk < nblk; ++blk) {
    const int k0 = blk * 32;
    __syncthreads();
    if (wu == 0) {    // TDM: 32 keys x 128 hd, contiguous rows, pad to stride 136
      tdm_load_2d_bf16(Kb + (size_t)k0 * HD_, Ks_off, /*tile_w=*/HD_, /*tile_h=*/32,
                       (unsigned long)HD_, (unsigned long)HD_,
                       /*pad 64DW*/5, /*+4DW*/3);
    }
    {   // V tile -> LDS transposed, pair-packed
      union { uint4 q[2]; unsigned short s[16]; } lo, hi;
      const uint4* pl = (const uint4*)(Vb + (size_t)(k0 + vkey2) * HD_ + vc0);
      const uint4* ph = (const uint4*)(Vb + (size_t)(k0 + vkey2 + 1) * HD_ + vc0);
      lo.q[0] = pl[0]; lo.q[1] = pl[1];
      hi.q[0] = ph[0]; hi.q[1] = ph[1];
#pragma unroll
      for (int e = 0; e < 16; ++e)
        *(unsigned*)(Vt + (vc0 + e) * 40 + vkey2) =
            (unsigned)lo.s[e] | ((unsigned)hi.s[e] << 16);
    }
    if (wu == 0) __builtin_amdgcn_s_wait_tensorcnt(0);
    __syncthreads();

    // scores S = Q * K^T : two 16x16 N-tiles, K-dim = HD (4 WMMAs each)
    v8f sa[2];
    sa[0] = vzero8(); sa[1] = vzero8();
#pragma unroll
    for (int nt = 0; nt < 2; ++nt)
#pragma unroll
      for (int f = 0; f < 4; ++f) {
        Frag16 bfr;   // B_t[key][hd] == Ks natural layout
#pragma unroll
        for (int j = 0; j < 8; ++j)
          bfr.u[j] = *(const unsigned*)(Ks + (nt * 16 + row) * 136 + f * 32 + kpat(j, half));
        sa[nt] = __builtin_amdgcn_wmma_f32_16x16x32_bf16(
            false, qf[f].v, false, bfr.v, (short)0, sa[nt], false, false);
      }

    // mask + online softmax (row stats per (e, half): m = e + 8*half, n = lane&15)
    const int ki0 = k0 + row, ki1 = k0 + 16 + row;
    const bool am0 = amask[b * S_ + ki0] > 0;
    const bool am1 = amask[b * S_ + ki1] > 0;
    float bm[8], alpha[8], ps[8], pv0[8], pv1[8];
#pragma unroll
    for (int e = 0; e < 8; ++e) {
      const int qi = qbase + e + 8 * half;
      const float s0 = (am0 && ki0 <= qi) ? sa[0][e] * scale : NEGBIG;
      const float s1 = (am1 && ki1 <= qi) ? sa[1][e] * scale : NEGBIG;
      pv0[e] = s0; pv1[e] = s1;
      bm[e] = fmaxf(s0, s1);
    }
#pragma unroll
    for (int m = 1; m < 16; m <<= 1)
#pragma unroll
      for (int e = 0; e < 8; ++e) bm[e] = fmaxf(bm[e], __shfl_xor(bm[e], m, 32));
#pragma unroll
    for (int e = 0; e < 8; ++e) {
      const float mn = fmaxf(mrow[e], bm[e]);
      alpha[e] = (mn <= 0.5f * NEGBIG) ? 1.0f : __expf(mrow[e] - mn);
      mrow[e] = mn;
      const float p0 = (pv0[e] <= 0.5f * NEGBIG) ? 0.f : __expf(pv0[e] - mn);
      const float p1 = (pv1[e] <= 0.5f * NEGBIG) ? 0.f : __expf(pv1[e] - mn);
      pv0[e] = p0; pv1[e] = p1; ps[e] = p0 + p1;
    }
#pragma unroll
    for (int m = 1; m < 16; m <<= 1)
#pragma unroll
      for (int e = 0; e < 8; ++e) ps[e] += __shfl_xor(ps[e], m, 32);
#pragma unroll
    for (int e = 0; e < 8; ++e) lrow[e] = lrow[e] * alpha[e] + ps[e];
#pragma unroll
    for (int t = 0; t < 8; ++t)
#pragma unroll
      for (int e = 0; e < 8; ++e) oacc[t][e] *= alpha[e];

    // restage P (C-layout -> A-layout) through per-wave LDS
    unsigned short* pw = Ps + w * (16 * 40);
#pragma unroll
    for (int e = 0; e < 8; ++e) {
      pw[(e + 8 * half) * 40 + row] = f2bf(pv0[e]);
      pw[(e + 8 * half) * 40 + 16 + row] = f2bf(pv1[e]);
    }
    Frag16 pf;
#pragma unroll
    for (int j = 0; j < 8; ++j)
      pf.u[j] = *(const unsigned*)(pw + row * 40 + kpat(j, half));

    // O += P * V : 8 hd tiles, K-dim = 32 keys; B_t[hd][key] pair reads
#pragma unroll
    for (int t = 0; t < 8; ++t) {
      Frag16 vf;
#pragma unroll
      for (int j = 0; j < 8; ++j)
        vf.u[j] = *(const unsigned*)(Vt + (t * 16 + row) * 40 + kpat(j, half));
      oacc[t] = __builtin_amdgcn_wmma_f32_16x16x32_bf16(
          false, pf.v, false, vf.v, (short)0, oacc[t], false, false);
    }
  }

#pragma unroll
  for (int e = 0; e < 8; ++e) lrow[e] = 1.0f / lrow[e];
#pragma unroll
  for (int t = 0; t < 8; ++t)
#pragma unroll
    for (int e = 0; e < 8; ++e) {
      const int qi = qbase + e + 8 * half;
      ctx[((size_t)b * S_ + qi) * H_ + h * HD_ + t * 16 + row] =
          f2bf(oacc[t][e] * lrow[e]);
    }
}

// ---------------- host launcher ----------------
extern "C" void kernel_launch(void* const* d_in, const int* in_sizes, int n_in,
                              void* d_out, int out_size, void* d_ws, size_t ws_size,
                              hipStream_t stream) {
  (void)in_sizes; (void)n_in; (void)out_size; (void)ws_size;
  const float* hidden  = (const float*)d_in[0];
  const float* w_qkv   = (const float*)d_in[1];
  const float* w_dense = (const float*)d_in[2];
  const float* ln_a_s  = (const float*)d_in[3];
  const float* ln_a_b  = (const float*)d_in[4];
  const float* ln_m_s  = (const float*)d_in[5];
  const float* ln_m_b  = (const float*)d_in[6];
  const float* w_fc    = (const float*)d_in[7];
  const float* w_out   = (const float*)d_in[8];
  const float* sin_tab = (const float*)d_in[9];
  const float* cos_tab = (const float*)d_in[10];
  const int*   amask   = (const int*)d_in[11];
  const int*   pos_ids = (const int*)d_in[12];
  float* out = (float*)d_out;

  char* ws = (char*)d_ws;
  size_t off = 0;
  auto alloc = [&](size_t bytes) -> char* {
    char* p = ws + off;
    off += (bytes + 255) & ~(size_t)255;
    return p;
  };
  unsigned short* wqkv_b   = (unsigned short*)alloc((size_t)H_ * QKV_ * 2);
  unsigned short* wdense_b = (unsigned short*)alloc((size_t)H_ * H_ * 2);
  unsigned short* wfc_b    = (unsigned short*)alloc((size_t)H_ * FF_ * 2);
  unsigned short* wout_b   = (unsigned short*)alloc((size_t)FF_ * H_ * 2);
  unsigned short* attn_in  = (unsigned short*)alloc((size_t)M2_ * H_ * 2);
  unsigned short* mlp_in   = (unsigned short*)alloc((size_t)M2_ * H_ * 2);
  float*          qkv_f    = (float*)alloc((size_t)M2_ * QKV_ * 4);
  unsigned short* Qb       = (unsigned short*)alloc((size_t)B_ * NH_ * S_ * HD_ * 2);
  unsigned short* Kb       = (unsigned short*)alloc((size_t)B_ * NKV_ * S_ * HD_ * 2);
  unsigned short* Vb       = (unsigned short*)alloc((size_t)B_ * NKV_ * S_ * HD_ * 2);
  unsigned short* ctxb     = (unsigned short*)alloc((size_t)M2_ * H_ * 2);
  float*          attn_out = (float*)alloc((size_t)M2_ * H_ * 4);
  unsigned short* fcact_b  = (unsigned short*)alloc((size_t)M2_ * FF_ * 2);

  // 1) weight conversions to bf16
  convert_bf16_kernel<<<2048, 256, 0, stream>>>(w_qkv, wqkv_b, (long)H_ * QKV_ / 4);
  convert_bf16_kernel<<<2048, 256, 0, stream>>>(w_dense, wdense_b, (long)H_ * H_ / 4);
  convert_bf16_kernel<<<4096, 256, 0, stream>>>(w_fc, wfc_b, (long)H_ * FF_ / 4);
  convert_bf16_kernel<<<4096, 256, 0, stream>>>(w_out, wout_b, (long)FF_ * H_ / 4);

  // 2) parallel layernorms
  ln_dual_kernel<<<M2_, 256, 0, stream>>>(hidden, ln_a_s, ln_a_b, ln_m_s, ln_m_b,
                                          attn_in, mlp_in);

  // 3) QKV projection (bf16 WMMA, f32 out)
  gemm_bf16_kernel<0><<<dim3(QKV_ / 128, M2_ / 128), 256, 0, stream>>>(
      attn_in, wqkv_b, QKV_, H_, qkv_f, nullptr, nullptr, nullptr);

  // 4) RoPE + grouped head split -> bf16 Q/K/V
  rope_split_kernel<<<B_ * S_, 128, 0, stream>>>(qkv_f, sin_tab, cos_tab, pos_ids,
                                                 Qb, Kb, Vb);

  // 5) flash attention -> bf16 ctx [B*S, H]
  flash_attn_kernel<<<dim3(B_ * NH_, S_ / 64), 128, 0, stream>>>(Qb, Kb, Vb, amask, ctxb);

  // 6) dense projection -> f32 attn_out
  gemm_bf16_kernel<0><<<dim3(H_ / 128, M2_ / 128), 256, 0, stream>>>(
      ctxb, wdense_b, H_, H_, attn_out, nullptr, nullptr, nullptr);

  // 7) FC + fused exact GELU -> bf16
  gemm_bf16_kernel<1><<<dim3(FF_ / 128, M2_ / 128), 256, 0, stream>>>(
      mlp_in, wfc_b, FF_, H_, nullptr, fcact_b, nullptr, nullptr);

  // 8) out projection + fused parallel residual (hidden + attn + mlp) -> d_out
  gemm_bf16_kernel<2><<<dim3(H_ / 128, M2_ / 128), 256, 0, stream>>>(
      fcact_b, wout_b, H_, FF_, out, nullptr, hidden, attn_out);
}
